// _RPN_79551384256545
// MI455X (gfx1250) — matrix operand order repaired and checked by reading, not compiled, not run on hardware
//
#include <hip/hip_runtime.h>
#include <cstdint>
#include <cstddef>
#include <math.h>

// ---------------- constants ----------------
#define NB    2        // batch
#define HW    2500     // 50*50
#define NA    22500    // anchors per batch
#define PRE   4000
#define POST  2000
#define SORTN 32768
#define NMSTH 0.7f

// ---------------- types ----------------
typedef _Float16 h16v __attribute__((ext_vector_type(16)));
typedef _Float16 h8v  __attribute__((ext_vector_type(8)));
typedef float    f8v  __attribute__((ext_vector_type(8)));
typedef float    f4v  __attribute__((ext_vector_type(4)));
typedef int      v4i_vs __attribute__((vector_size(16)));   // matches builtin param

#define AS1 __attribute__((address_space(1)))
#define AS3 __attribute__((address_space(3)))

__device__ __forceinline__ int imin(int a, int b) { return a < b ? a : b; }
__device__ __forceinline__ int imax(int a, int b) { return a > b ? a : b; }

// 16-byte global -> LDS copy; async CDNA5 path if the builtin is available.
__device__ __forceinline__ void cp16(_Float16* lds, const _Float16* g) {
#if defined(__AMDGCN__) && __has_builtin(__builtin_amdgcn_global_load_async_to_lds_b128)
  __builtin_amdgcn_global_load_async_to_lds_b128(
      (AS1 v4i_vs*)(void*)g, (AS3 v4i_vs*)(void*)lds, 0, 0);
#else
  *(f4v*)lds = *(const f4v*)g;
#endif
}
__device__ __forceinline__ void cp_wait() {
#if defined(__AMDGCN__) && __has_builtin(__builtin_amdgcn_s_wait_asynccnt)
  __builtin_amdgcn_s_wait_asynccnt(0);
#endif
}

__device__ __forceinline__ float urandf(unsigned x) {
  x ^= x >> 17; x *= 0xed5ad4bbu; x ^= x >> 11; x *= 0xac4c1b51u;
  x ^= x >> 15; x *= 0x31848babu; x ^= x >> 14;
  return (float)(x >> 8) * (1.0f / 16777216.0f);
}

__device__ __forceinline__ float iou1(const float* a, const float* b) {
  float aa = (a[2]-a[0]+1.f)*(a[3]-a[1]+1.f);
  float ab = (b[2]-b[0]+1.f)*(b[3]-b[1]+1.f);
  float ix1 = fmaxf(a[0], b[0]), iy1 = fmaxf(a[1], b[1]);
  float ix2 = fminf(a[2], b[2]), iy2 = fminf(a[3], b[3]);
  float iw = fmaxf(ix2-ix1+1.f, 0.f), ih = fmaxf(iy2-iy1+1.f, 0.f);
  float inter = iw*ih, uni = aa+ab-inter;
  return uni > 0.f ? inter/uni : 0.f;
}

// ---------------- packing ----------------
// NCHW f32 -> padded (52x52) NHWC f16
__global__ void rpn_pack_a(const float* __restrict__ bf, _Float16* __restrict__ apad) {
  int t = blockIdx.x * blockDim.x + threadIdx.x;
  if (t >= NB*512*50*50) return;
  int tmp = t;
  int x = tmp % 50; tmp /= 50;
  int y = tmp % 50; tmp /= 50;
  int c = tmp % 512; int b = tmp / 512;
  apad[(((size_t)b*52 + (y+1))*52 + (x+1))*512 + c] = (_Float16)bf[t];
}

// (O,C,3,3) f32 -> [tap][o][c] f16
__global__ void rpn_pack_w(const float* __restrict__ w, _Float16* __restrict__ wt) {
  int t = blockIdx.x * blockDim.x + threadIdx.x;
  if (t >= 512*512*9) return;
  int tmp = t;
  int dx = tmp % 3; tmp /= 3;
  int dy = tmp % 3; tmp /= 3;
  int c = tmp % 512; int o = tmp / 512;
  wt[((size_t)(dy*3+dx)*512 + o)*512 + c] = (_Float16)w[t];
}

// ---------------- 3x3 conv as implicit GEMM on WMMA ----------------
// grid (79, 2, NB), block 256 = 8 waves; wave (wm,wn) in 2x4 -> 32x256 tile/WG.
__global__ __launch_bounds__(256)
void rpn_conv_wmma(const _Float16* __restrict__ apad, const _Float16* __restrict__ wt,
                   const float* __restrict__ bias, float* __restrict__ rpn1) {
  __shared__ __align__(16) _Float16 sA[32*40];
  __shared__ __align__(16) _Float16 sB[256*40];
  const int bx = blockIdx.x, by = blockIdx.y, b = blockIdx.z;
  const int tid = threadIdx.x;
  const int lane = tid & 31, wave = tid >> 5;
  const int wm = wave >> 2, wn = wave & 3;
  const int lo = lane & 15, hi = lane >> 4;
  const int m_base = bx * 32;

  f8v acc[4];
#pragma unroll
  for (int j = 0; j < 4; j++)
#pragma unroll
    for (int v = 0; v < 8; v++) acc[j][v] = 0.f;

  for (int t9 = 0; t9 < 9; ++t9) {
    const int dy = t9 / 3, dx = t9 % 3;
    const _Float16* wsrc = wt + ((size_t)t9*512 + by*256)*512;
    for (int c0 = 0; c0 < 512; c0 += 32) {
      // ---- stage A (32 rows x 32 K halves) ----
      if (tid < 128) {
        int r = tid >> 2, seg = tid & 3;
        int m = m_base + r;
        _Float16* dst = &sA[r*40 + seg*8];
        if (m < HW) {
          int y = m / 50, x = m % 50;
          const _Float16* src =
              apad + (((size_t)b*52 + (y+dy))*52 + (x+dx))*512 + c0 + seg*8;
          cp16(dst, src);
        } else {
#pragma unroll
          for (int q = 0; q < 8; q++) dst[q] = (_Float16)0.f;
        }
      }
      // ---- stage B (256 N rows x 32 K halves, N-major for column reads) ----
      {
        const _Float16* src = wsrc + (size_t)tid*512 + c0;
        _Float16* dst = &sB[tid*40];
        cp16(dst,      src);
        cp16(dst + 8,  src + 8);
        cp16(dst + 16, src + 16);
        cp16(dst + 24, src + 24);
        if (c0 + 32 < 512) __builtin_prefetch(src + 32, 0, 1);
      }
      cp_wait();
      __syncthreads();

      // ---- A fragment: lane lo holds row M=lo; hi selects K octets ----
      h16v afrag;
      {
        const _Float16* ap = &sA[(wm*16 + lo)*40 + hi*8];
        h8v a0 = *(const h8v*)ap;
        h8v a1 = *(const h8v*)(ap + 16);
#pragma unroll
        for (int q = 0; q < 8; q++) { afrag[q] = a0[q]; afrag[8+q] = a1[q]; }
      }
#pragma unroll
      for (int j = 0; j < 4; j++) {
        const _Float16* bp = &sB[(wn*64 + j*16 + lo)*40 + hi*8];
        h8v b0 = *(const h8v*)bp;
        h8v b1 = *(const h8v*)(bp + 16);
        h16v bfrag;
#pragma unroll
        for (int q = 0; q < 8; q++) { bfrag[q] = b0[q]; bfrag[8+q] = b1[q]; }
        acc[j] = __builtin_amdgcn_wmma_f32_16x16x32_f16(
            false, afrag, false, bfrag, (short)0, acc[j], false, false);
      }
      __syncthreads();
    }
  }
  // ---- epilogue: D layout M = v + 8*hi, N = lane&15 ----
#pragma unroll
  for (int j = 0; j < 4; j++) {
    int n = by*256 + wn*64 + j*16 + lo;
    float bv = bias[n];
#pragma unroll
    for (int v = 0; v < 8; v++) {
      int m = m_base + wm*16 + v + 8*hi;
      if (m < HW) {
        float r = acc[j][v] + bv;
        rpn1[((size_t)b*HW + m)*512 + n] = r > 0.f ? r : 0.f;
      }
    }
  }
}

// ---------------- meta-attention MLP chain ----------------
__global__ __launch_bounds__(1024)
void rpn_meta_a(const float* att, const float* fga, const float* ma_w, const float* ma_b,
                const float* fgm_w1, const float* fgm_b1, float* ma, float* h) {
  __shared__ float fm[512];
  int tid = threadIdx.x;
  if (tid < 512) {
    float s = 0.f;
    for (int g = 0; g < 20; g++) s += fga[g*512 + tid];
    fm[tid] = s * (1.f/20.f);
  }
  __syncthreads();
  { float s = 0.f; const float* w = ma_w + (size_t)tid*2048;
    for (int i = 0; i < 2048; i++) s += att[i]*w[i];
    ma[tid] = 1.f/(1.f + expf(-(s + ma_b[tid]))); }
  { float s = 0.f; const float* w = fgm_w1 + (size_t)tid*512;
    for (int i = 0; i < 512; i++) s += fm[i]*w[i];
    s += fgm_b1[tid]; h[tid] = s > 0.f ? s : 0.f; }
}
__global__ __launch_bounds__(1024)
void rpn_meta_b(const float* h, const float* w2, const float* b2, float* mfa) {
  int tid = threadIdx.x;
  float s = 0.f; const float* w = w2 + (size_t)tid*1024;
  for (int i = 0; i < 1024; i++) s += h[i]*w[i];
  mfa[tid] = 1.f/(1.f + expf(-(s + b2[tid])));
}
__global__ __launch_bounds__(1024)
void rpn_meta_c(const float* ma, const float* mfa, const float* fw, const float* fb, float* fused) {
  int tid = threadIdx.x;
  float s = 0.f; const float* w = fw + (size_t)tid*2048;
  for (int i = 0; i < 1024; i++) s += ma[i]*w[i];
  for (int i = 0; i < 1024; i++) s += mfa[i]*w[1024 + i];
  s += fb[tid]; fused[tid] = s > 0.f ? s : 0.f;
}
__global__ void rpn_meta_d(const float* fused, const float* w1, const float* b1, float* h1) {
  int tid = threadIdx.x;
  float s = 0.f; const float* w = w1 + (size_t)tid*1024;
  for (int i = 0; i < 1024; i++) s += fused[i]*w[i];
  s += b1[tid]; h1[tid] = s > 0.f ? s : 0.f;
}
__global__ void rpn_meta_e(const float* h1, const float* w2, const float* b2, float* h2) {
  int tid = threadIdx.x;
  float s = 0.f; const float* w = w2 + (size_t)tid*512;
  for (int i = 0; i < 512; i++) s += h1[i]*w[i];
  s += b2[tid]; h2[tid] = s > 0.f ? s : 0.f;
}
__global__ void rpn_meta_f(const float* h2, const float* w3, const float* b3, float* dynw) {
  int o = blockIdx.x*blockDim.x + threadIdx.x;
  if (o >= 9216) return;
  float s = 0.f; const float* w = w3 + (size_t)o*128;
  for (int i = 0; i < 128; i++) s += h2[i]*w[i];
  dynw[o] = s + b3[o];
}

// ---------------- 1x1 heads ----------------
__global__ void rpn_heads(const float* __restrict__ rpn1, const float* __restrict__ cls_w,
                          const float* __restrict__ cls_b, const float* __restrict__ bbox_w,
                          const float* __restrict__ bbox_b, const float* __restrict__ dynw,
                          float* __restrict__ clsS, float* __restrict__ bbox) {
  int pos = blockIdx.x;              // 0..4999
  int b = pos / HW, m = pos % HW;
  __shared__ float sf[512];
  const float* row = rpn1 + ((size_t)b*HW + m)*512;
  for (int i = threadIdx.x; i < 512; i += blockDim.x) sf[i] = row[i];
  __syncthreads();
  int o = threadIdx.x;
  if (o < 18) {
    float dc = 0.f, da = 0.f;
    const float* wc = cls_w + (size_t)o*512;
    const float* wa = dynw  + (size_t)o*512;
    for (int i = 0; i < 512; i++) { float f = sf[i]; dc += f*wc[i]; da += f*wa[i]; }
    dc += cls_b[o];
    clsS[((size_t)b*HW + m)*18 + o] = dc*da + dc;
  } else if (o < 54) {
    int q = o - 18;
    float d = 0.f; const float* wb = bbox_w + (size_t)q*512;
    for (int i = 0; i < 512; i++) d += sf[i]*wb[i];
    bbox[((size_t)b*HW + m)*36 + q] = d + bbox_b[q];
  }
}

// ---------------- anchors / proposals ----------------
__global__ void rpn_anchors(float* __restrict__ anch) {
  int n = blockIdx.x*blockDim.x + threadIdx.x;
  if (n >= NA) return;
  int a = n % 9, m = n / 9, x = m % 50, y = m / 50;
  int r = a / 3, s = a % 3;
  const float ratios[3] = {0.5f, 1.f, 2.f};
  const float scales[3] = {8.f, 16.f, 32.f};
  float ws = rintf(sqrtf(256.f/ratios[r]));
  float hs = rintf(ws*ratios[r]);
  float x1 = 7.5f - 0.5f*(ws-1.f), y1 = 7.5f - 0.5f*(hs-1.f);
  float x2 = 7.5f + 0.5f*(ws-1.f), y2 = 7.5f + 0.5f*(hs-1.f);
  float w2 = x2-x1+1.f, h2 = y2-y1+1.f;
  float xc = x1 + 0.5f*(w2-1.f), yc = y1 + 0.5f*(h2-1.f);
  float wss = w2*scales[s], hss = h2*scales[s];
  float sx = x*16.f, sy = y*16.f;
  anch[n*4+0] = sx + xc - 0.5f*(wss-1.f);
  anch[n*4+1] = sy + yc - 0.5f*(hss-1.f);
  anch[n*4+2] = sx + xc + 0.5f*(wss-1.f);
  anch[n*4+3] = sy + yc + 0.5f*(hss-1.f);
}

__global__ void rpn_props(const float* __restrict__ clsS, const float* __restrict__ bbox,
                          const float* __restrict__ anch, const float* __restrict__ im_info,
                          float* __restrict__ score, float* __restrict__ props) {
  int t = blockIdx.x*blockDim.x + threadIdx.x;
  if (t >= NB*NA) return;
  int b = t / NA, n = t % NA, m = n / 9, a = n % 9;
  const float* cr = clsS + ((size_t)b*HW + m)*18;
  float s0 = cr[a], s1 = cr[9 + a];
  score[t] = 1.f/(1.f + expf(s0 - s1));
  const float* dr = bbox + ((size_t)b*HW + m)*36 + a*4;
  const float* an = anch + (size_t)n*4;
  float aw = an[2]-an[0]+1.f, ah = an[3]-an[1]+1.f;
  float acx = an[0] + 0.5f*aw, acy = an[1] + 0.5f*ah;
  float pcx = dr[0]*aw + acx, pcy = dr[1]*ah + acy;
  float pw = expf(dr[2])*aw, ph = expf(dr[3])*ah;
  float imh = im_info[b*3+0] - 1.f, imw = im_info[b*3+1] - 1.f;
  float* o = props + (size_t)t*4;
  o[0] = fminf(fmaxf(pcx - 0.5f*pw, 0.f), imw);
  o[1] = fminf(fmaxf(pcy - 0.5f*ph, 0.f), imh);
  o[2] = fminf(fmaxf(pcx + 0.5f*pw, 0.f), imw);
  o[3] = fminf(fmaxf(pcy + 0.5f*ph, 0.f), imh);
}

// ---------------- bitonic sort (descending, tie-break by lower idx) ----------------
__global__ void rpn_sortinit_scores(const float* __restrict__ score,
                                    float* __restrict__ key, int* __restrict__ idx) {
  int t = blockIdx.x*blockDim.x + threadIdx.x;
  if (t >= NB*SORTN) return;
  int b = t / SORTN, i = t % SORTN;
  key[t] = (i < NA) ? score[b*NA + i] : -1e30f;
  idx[t] = i;
}
__global__ void rpn_sortinit_flag(const float* __restrict__ labels, float want, unsigned salt,
                                  float* __restrict__ key, int* __restrict__ idx) {
  int t = blockIdx.x*blockDim.x + threadIdx.x;
  if (t >= NB*SORTN) return;
  int b = t / SORTN, i = t % SORTN;
  float v = -1e30f;
  if (i < NA && labels[b*NA + i] == want)
    v = urandf(((unsigned)(b*SORTN + i))*2654435761u + salt);
  key[t] = v; idx[t] = i;
}
__global__ void rpn_bitonic_global(float* key, int* idx, int j, int k) {
  int i = blockIdx.x*blockDim.x + threadIdx.x;
  int b = blockIdx.y;
  float* kb = key + (size_t)b*SORTN; int* ib = idx + (size_t)b*SORTN;
  int l = i ^ j;
  if (l <= i) return;
  float ki = kb[i], kl = kb[l]; int ii = ib[i], il = ib[l];
  bool up = ((i & k) == 0);
  bool l_first = (kl > ki) || (kl == ki && il < ii);
  if (l_first == up) { kb[i] = kl; kb[l] = ki; ib[i] = il; ib[l] = ii; }
}
__global__ __launch_bounds__(1024)
void rpn_bitonic_lds(float* key, int* idx, int j0, int k) {
  __shared__ float sk[4096]; __shared__ int si[4096];
  int b = blockIdx.y; int base = blockIdx.x*4096;
  float* kb = key + (size_t)b*SORTN + base; int* ib = idx + (size_t)b*SORTN + base;
  for (int q = threadIdx.x; q < 4096; q += 1024) { sk[q] = kb[q]; si[q] = ib[q]; }
  __syncthreads();
  for (int j = j0; j >= 1; j >>= 1) {
    for (int p = 0; p < 2; p++) {
      int t = threadIdx.x + p*1024;
      int i = ((t & ~(j-1)) << 1) | (t & (j-1));
      int l = i | j;
      bool up = (((base + i) & k) == 0);
      float ki = sk[i], kl = sk[l]; int ii = si[i], il = si[l];
      bool l_first = (kl > ki) || (kl == ki && il < ii);
      if (l_first == up) { sk[i] = kl; sk[l] = ki; si[i] = il; si[l] = ii; }
    }
    __syncthreads();
  }
  for (int q = threadIdx.x; q < 4096; q += 1024) { kb[q] = sk[q]; ib[q] = si[q]; }
}
__global__ void rpn_rank(const int* __restrict__ idx, int* __restrict__ rank) {
  int t = blockIdx.x*blockDim.x + threadIdx.x;
  if (t >= NB*SORTN) return;
  int b = t / SORTN, pos = t % SORTN;
  int id = idx[t];
  if (id < NA) rank[b*NA + id] = pos;
}
__global__ void rpn_gather_top(const float* __restrict__ props, const int* __restrict__ idx,
                               float* __restrict__ topb) {
  int t = blockIdx.x*blockDim.x + threadIdx.x;
  if (t >= NB*PRE) return;
  int b = t / PRE, i = t % PRE;
  int id = idx[(size_t)b*SORTN + i];
  const float* s = props + ((size_t)b*NA + id)*4;
  float* d = topb + (size_t)t*4;
  d[0] = s[0]; d[1] = s[1]; d[2] = s[2]; d[3] = s[3];
}

// ---------------- NMS ----------------
#define NMSW 63   // ceil(4000/64)
__global__ void rpn_nms_mask(const float* __restrict__ topb, unsigned long long* __restrict__ mask) {
  int t = blockIdx.x*blockDim.x + threadIdx.x;
  if (t >= NB*PRE*NMSW) return;
  int b = t / (PRE*NMSW); int r = t % (PRE*NMSW);
  int i = r / NMSW, c = r % NMSW;
  const float* bp = topb + (size_t)b*PRE*4;
  float bi[4] = {bp[i*4], bp[i*4+1], bp[i*4+2], bp[i*4+3]};
  unsigned long long mb = 0ull;
  int j0 = c*64;
  for (int jj = 0; jj < 64; jj++) {
    int j = j0 + jj;
    if (j <= i || j >= PRE) continue;
    if (iou1(bi, bp + j*4) > NMSTH) mb |= (1ull << jj);
  }
  mask[((size_t)b*PRE + i)*NMSW + c] = mb;
}
__global__ void rpn_nms_scan(const unsigned long long* __restrict__ mask,
                             int* __restrict__ keepidx, int* __restrict__ keepcnt) {
  int b = blockIdx.x;
  __shared__ unsigned long long remv[NMSW];
  __shared__ int alive_s;
  __shared__ int cnt_s;
  if (threadIdx.x < NMSW) remv[threadIdx.x] = 0ull;
  if (threadIdx.x == 0) cnt_s = 0;
  __syncthreads();
  for (int i = 0; i < PRE; i++) {
    if (threadIdx.x == 0) alive_s = !((remv[i >> 6] >> (i & 63)) & 1ull);
    __syncthreads();
    if (alive_s) {
      if (threadIdx.x < NMSW)
        remv[threadIdx.x] |= mask[((size_t)b*PRE + i)*NMSW + threadIdx.x];
      if (threadIdx.x == 0) {
        int p = cnt_s++;
        if (p < POST) keepidx[b*POST + p] = i;
      }
    }
    __syncthreads();
  }
  if (threadIdx.x == 0) keepcnt[b] = cnt_s;
}
__global__ void rpn_rois(const float* __restrict__ topb, const int* __restrict__ keepidx,
                         const int* __restrict__ keepcnt, float* __restrict__ out) {
  int t = blockIdx.x*blockDim.x + threadIdx.x;
  if (t >= NB*POST) return;
  int b = t / POST, p = t % POST;
  float* o = out + (size_t)t*5;
  o[0] = (float)b;
  int cnt = imin(keepcnt[b], POST);
  if (p < cnt) {
    int i = keepidx[b*POST + p];
    const float* bx = topb + ((size_t)b*PRE + i)*4;
    o[1] = bx[0]; o[2] = bx[1]; o[3] = bx[2]; o[4] = bx[3];
  } else {
    o[1] = 0.f; o[2] = 0.f; o[3] = 0.f; o[4] = 0.f;
  }
}

// ---------------- anchor target ----------------
__global__ void rpn_iou_at(const float* __restrict__ anch, const float* __restrict__ gt,
                           const int* __restrict__ numb, const float* __restrict__ im_info,
                           float* __restrict__ ov) {
  int t = blockIdx.x*blockDim.x + threadIdx.x;
  if (t >= NB*NA*20) return;
  int b = t / (NA*20); int r = t % (NA*20);
  int n = r / 20, g = r % 20;
  const float* an = anch + (size_t)n*4;
  float imh = im_info[b*3+0], imw = im_info[b*3+1];
  bool inside = (an[0] >= 0.f) && (an[1] >= 0.f) && (an[2] < imw) && (an[3] < imh);
  bool gvalid = g < numb[b];
  const float* gb = gt + ((size_t)b*20 + g)*5;
  float v = iou1(an, gb);
  ov[t] = v * (inside ? 1.f : 0.f) * (gvalid ? 1.f : 0.f);
}
__global__ void rpn_gtmax(const float* __restrict__ ov, float* __restrict__ gtmax) {
  int b = blockIdx.x / 20, g = blockIdx.x % 20;
  __shared__ float sm[256];
  float mx = 0.f;
  for (int n = threadIdx.x; n < NA; n += 256)
    mx = fmaxf(mx, ov[((size_t)b*NA + n)*20 + g]);
  sm[threadIdx.x] = mx; __syncthreads();
  for (int s = 128; s > 0; s >>= 1) {
    if (threadIdx.x < s) sm[threadIdx.x] = fmaxf(sm[threadIdx.x], sm[threadIdx.x + s]);
    __syncthreads();
  }
  if (threadIdx.x == 0) gtmax[b*20 + g] = (sm[0] <= 0.f) ? 100000.f : sm[0];
}
__global__ void rpn_labels(const float* __restrict__ ov, const float* __restrict__ gtmax,
                           const float* __restrict__ anch, const float* __restrict__ im_info,
                           float* __restrict__ labels, int* __restrict__ amax,
                           int* __restrict__ cntfg) {
  int t = blockIdx.x*blockDim.x + threadIdx.x;
  if (t >= NB*NA) return;
  int b = t / NA, n = t % NA;
  float mo = -1.f; int ag = 0; bool fgg = false;
  const float* ovr = ov + (size_t)t*20;
  const float* gm = gtmax + b*20;
  for (int g = 0; g < 20; g++) {
    float v = ovr[g];
    if (v > mo) { mo = v; ag = g; }
    if (v == gm[g]) fgg = true;
  }
  const float* an = anch + (size_t)n*4;
  float imh = im_info[b*3+0], imw = im_info[b*3+1];
  bool inside = (an[0] >= 0.f) && (an[1] >= 0.f) && (an[2] < imw) && (an[3] < imh);
  float l = -1.f;
  if (inside && mo < 0.3f) l = 0.f;
  if (fgg) l = 1.f;
  if (inside && mo >= 0.7f) l = 1.f;
  labels[t] = l; amax[t] = ag;
  if (l == 1.f) atomicAdd(&cntfg[b], 1);
}
__global__ void rpn_labels2(float* __restrict__ labels, const int* __restrict__ rankfg,
                            const int* __restrict__ rankbg, const int* __restrict__ cntfg,
                            int* __restrict__ cntex) {
  int t = blockIdx.x*blockDim.x + threadIdx.x;
  if (t >= NB*NA) return;
  int b = t / NA;
  float l = labels[t];
  if (l == 1.f && rankfg[t] >= 128) l = -1.f;
  int budget = 256 - imin(cntfg[b], 128);
  if (l == 0.f && rankbg[t] >= budget) l = -1.f;
  labels[t] = l;
  if (l >= 0.f) atomicAdd(&cntex[b], 1);
}
__global__ void rpn_targets(const float* __restrict__ anch, const float* __restrict__ gt,
                            const int* __restrict__ amax, float* __restrict__ tgt) {
  int t = blockIdx.x*blockDim.x + threadIdx.x;
  if (t >= NB*NA) return;
  int b = t / NA, n = t % NA;
  const float* gb = gt + ((size_t)b*20 + amax[t])*5;
  const float* an = anch + (size_t)n*4;
  float aw = an[2]-an[0]+1.f, ah = an[3]-an[1]+1.f;
  float acx = an[0] + 0.5f*aw, acy = an[1] + 0.5f*ah;
  float gw = fmaxf(gb[2]-gb[0]+1.f, 1.f), gh = fmaxf(gb[3]-gb[1]+1.f, 1.f);
  float gcx = gb[0] + 0.5f*gw, gcy = gb[1] + 0.5f*gh;
  float* o = tgt + (size_t)t*4;
  o[0] = (gcx - acx)/aw; o[1] = (gcy - acy)/ah;
  o[2] = logf(gw/aw);    o[3] = logf(gh/ah);
}

// ---------------- losses ----------------
__global__ void rpn_loss_cls(const float* __restrict__ clsS, const float* __restrict__ labels,
                             float* __restrict__ acc) {
  __shared__ float sn[256], sd[256];
  int t = blockIdx.x*blockDim.x + threadIdx.x;
  float num = 0.f, den = 0.f;
  if (t < NB*NA) {
    int b = t / NA, n = t % NA, m = n / 9, a = n % 9;
    float l = labels[t];
    if (l != -1.f) {
      const float* cr = clsS + ((size_t)b*HW + m)*18;
      float s0 = cr[a], s1 = cr[9 + a];
      float mx = fmaxf(s0, s1);
      float lse = mx + logf(expf(s0 - mx) + expf(s1 - mx));
      num = lse - ((l > 0.5f) ? s1 : s0);
      den = 1.f;
    }
  }
  sn[threadIdx.x] = num; sd[threadIdx.x] = den; __syncthreads();
  for (int s = 128; s > 0; s >>= 1) {
    if (threadIdx.x < s) { sn[threadIdx.x] += sn[threadIdx.x+s]; sd[threadIdx.x] += sd[threadIdx.x+s]; }
    __syncthreads();
  }
  if (threadIdx.x == 0) { atomicAdd(&acc[0], sn[0]); atomicAdd(&acc[1], sd[0]); }
}
__global__ void rpn_loss_box(const float* __restrict__ bbox, const float* __restrict__ tgt,
                             const float* __restrict__ labels, const int* __restrict__ cntex,
                             float* __restrict__ acc) {
  __shared__ float ss[256];
  int t = blockIdx.x*blockDim.x + threadIdx.x;
  float s = 0.f;
  if (t < NB*NA) {
    int b = t / NA, n = t % NA, m = n / 9, a = n % 9;
    float l = labels[t];
    if (l >= 0.f) {
      float ow = 1.f/(float)imax(cntex[b], 1);
      float inw = (l == 1.f) ? 1.f : 0.f;
      const float* pr = bbox + ((size_t)b*HW + m)*36 + a*4;
      const float* tg = tgt + (size_t)t*4;
      for (int k2 = 0; k2 < 4; k2++) {
        float d = inw*(pr[k2] - tg[k2]);
        float ad = fabsf(d);
        float sl = (ad < 1.f/9.f) ? 0.5f*9.f*d*d : ad - 0.5f/9.f;
        s += ow*sl;
      }
    }
  }
  ss[threadIdx.x] = s; __syncthreads();
  for (int q = 128; q > 0; q >>= 1) {
    if (threadIdx.x < q) ss[threadIdx.x] += ss[threadIdx.x + q];
    __syncthreads();
  }
  if (threadIdx.x == 0) atomicAdd(&acc[2], ss[0]);
}
__global__ void rpn_finalize(const float* __restrict__ acc, float* __restrict__ out) {
  out[NB*POST*5 + 0] = acc[0]/acc[1];
  out[NB*POST*5 + 1] = acc[2]*(1.f/(float)NB);
}

// ---------------- host-side sort driver ----------------
static void run_sort(float* key, int* idx, hipStream_t stream) {
  for (int k = 2; k <= SORTN; k <<= 1) {
    int j = k >> 1;
    while (j > 2048) {
      rpn_bitonic_global<<<dim3(SORTN/256, NB), 256, 0, stream>>>(key, idx, j, k);
      j >>= 1;
    }
    rpn_bitonic_lds<<<dim3(SORTN/4096, NB), 1024, 0, stream>>>(key, idx, j, k);
  }
}

// ---------------- entry ----------------
extern "C" void kernel_launch(void* const* d_in, const int* in_sizes, int n_in,
                              void* d_out, int out_size, void* d_ws, size_t ws_size,
                              hipStream_t stream) {
  (void)in_sizes; (void)n_in; (void)out_size; (void)ws_size;
  const float* base_feat  = (const float*)d_in[0];
  const float* im_info    = (const float*)d_in[1];
  const float* gt_boxes   = (const float*)d_in[2];
  const int*   num_boxes  = (const int*)d_in[3];
  const float* attentions = (const float*)d_in[4];
  const float* fg_att     = (const float*)d_in[5];
  const float* conv_w     = (const float*)d_in[6];
  const float* conv_b     = (const float*)d_in[7];
  const float* cls_w      = (const float*)d_in[8];
  const float* cls_b      = (const float*)d_in[9];
  const float* bbox_w     = (const float*)d_in[10];
  const float* bbox_b     = (const float*)d_in[11];
  const float* ma_w       = (const float*)d_in[12];
  const float* ma_b       = (const float*)d_in[13];
  const float* fgm_w1     = (const float*)d_in[14];
  const float* fgm_b1     = (const float*)d_in[15];
  const float* fgm_w2     = (const float*)d_in[16];
  const float* fgm_b2     = (const float*)d_in[17];
  const float* fuse_w     = (const float*)d_in[18];
  const float* fuse_b     = (const float*)d_in[19];
  const float* meta_w1    = (const float*)d_in[20];
  const float* meta_b1    = (const float*)d_in[21];
  const float* meta_w2    = (const float*)d_in[22];
  const float* meta_b2    = (const float*)d_in[23];
  const float* meta_w3    = (const float*)d_in[24];
  const float* meta_b3    = (const float*)d_in[25];
  float* out = (float*)d_out;

  size_t off = 0;
  auto alloc = [&](size_t bytes) -> void* {
    void* p = (char*)d_ws + off;
    off += (bytes + 255) & ~(size_t)255;
    return p;
  };
  _Float16* apad = (_Float16*)alloc((size_t)NB*52*52*512*2);
  _Float16* wt9  = (_Float16*)alloc((size_t)9*512*512*2);
  float* rpn1  = (float*)alloc((size_t)NB*HW*512*4);
  float* clsS  = (float*)alloc((size_t)NB*HW*18*4);
  float* bbox  = (float*)alloc((size_t)NB*HW*36*4);
  float* anch  = (float*)alloc((size_t)NA*4*4);
  float* props = (float*)alloc((size_t)NB*NA*4*4);
  float* score = (float*)alloc((size_t)NB*NA*4);
  float* skey  = (float*)alloc((size_t)NB*SORTN*4);
  int*   sidx  = (int*)  alloc((size_t)NB*SORTN*4);
  float* topb  = (float*)alloc((size_t)NB*PRE*4*4);
  unsigned long long* nmask = (unsigned long long*)alloc((size_t)NB*PRE*NMSW*8);
  int*   keepidx = (int*)alloc((size_t)NB*POST*4);
  float* ov    = (float*)alloc((size_t)NB*NA*20*4);
  float* gtmax = (float*)alloc((size_t)NB*20*4);
  float* labels= (float*)alloc((size_t)NB*NA*4);
  int*   amax  = (int*)  alloc((size_t)NB*NA*4);
  float* tgt   = (float*)alloc((size_t)NB*NA*4*4);
  int* rankfg  = (int*)alloc((size_t)NB*NA*4);
  int* rankbg  = (int*)alloc((size_t)NB*NA*4);
  float* mmA   = (float*)alloc(1024*4);
  float* mmH   = (float*)alloc(1024*4);
  float* mmFA  = (float*)alloc(1024*4);
  float* mmFU  = (float*)alloc(1024*4);
  float* mmH1  = (float*)alloc(512*4);
  float* mmH2  = (float*)alloc(128*4);
  float* dynw  = (float*)alloc(9216*4);
  float* accF  = (float*)alloc(256);
  int*   accI  = (int*)((char*)accF + 64);   // [0..1] cnt_fg, [2..3] cnt_ex, [4..5] keep_cnt

  (void)hipMemsetAsync(apad, 0, (size_t)NB*52*52*512*2, stream);
  (void)hipMemsetAsync(accF, 0, 256, stream);

  rpn_pack_a<<<dim3((NB*512*50*50 + 255)/256), 256, 0, stream>>>(base_feat, apad);
  rpn_pack_w<<<dim3((512*512*9 + 255)/256), 256, 0, stream>>>(conv_w, wt9);
  rpn_anchors<<<dim3((NA + 255)/256), 256, 0, stream>>>(anch);

  rpn_conv_wmma<<<dim3(79, 2, NB), 256, 0, stream>>>(apad, wt9, conv_b, rpn1);

  rpn_meta_a<<<1, 1024, 0, stream>>>(attentions, fg_att, ma_w, ma_b, fgm_w1, fgm_b1, mmA, mmH);
  rpn_meta_b<<<1, 1024, 0, stream>>>(mmH, fgm_w2, fgm_b2, mmFA);
  rpn_meta_c<<<1, 1024, 0, stream>>>(mmA, mmFA, fuse_w, fuse_b, mmFU);
  rpn_meta_d<<<1, 512, 0, stream>>>(mmFU, meta_w1, meta_b1, mmH1);
  rpn_meta_e<<<1, 128, 0, stream>>>(mmH1, meta_w2, meta_b2, mmH2);
  rpn_meta_f<<<dim3(36), 256, 0, stream>>>(mmH2, meta_w3, meta_b3, dynw);

  rpn_heads<<<dim3(NB*HW), 64, 0, stream>>>(rpn1, cls_w, cls_b, bbox_w, bbox_b, dynw, clsS, bbox);
  rpn_props<<<dim3((NB*NA + 255)/256), 256, 0, stream>>>(clsS, bbox, anch, im_info, score, props);

  rpn_sortinit_scores<<<dim3((NB*SORTN)/256), 256, 0, stream>>>(score, skey, sidx);
  run_sort(skey, sidx, stream);
  rpn_gather_top<<<dim3((NB*PRE + 255)/256), 256, 0, stream>>>(props, sidx, topb);

  rpn_nms_mask<<<dim3((NB*PRE*NMSW + 255)/256), 256, 0, stream>>>(topb, nmask);
  rpn_nms_scan<<<dim3(NB), 64, 0, stream>>>(nmask, keepidx, accI + 4);
  rpn_rois<<<dim3((NB*POST + 255)/256), 256, 0, stream>>>(topb, keepidx, accI + 4, out);

  rpn_iou_at<<<dim3((NB*NA*20 + 255)/256), 256, 0, stream>>>(anch, gt_boxes, num_boxes, im_info, ov);
  rpn_gtmax<<<dim3(NB*20), 256, 0, stream>>>(ov, gtmax);
  rpn_labels<<<dim3((NB*NA + 255)/256), 256, 0, stream>>>(ov, gtmax, anch, im_info, labels, amax, accI);

  rpn_sortinit_flag<<<dim3((NB*SORTN)/256), 256, 0, stream>>>(labels, 1.0f, 0x9e3779b9u, skey, sidx);
  run_sort(skey, sidx, stream);
  rpn_rank<<<dim3((NB*SORTN)/256), 256, 0, stream>>>(sidx, rankfg);
  rpn_sortinit_flag<<<dim3((NB*SORTN)/256), 256, 0, stream>>>(labels, 0.0f, 0x85ebca6bu, skey, sidx);
  run_sort(skey, sidx, stream);
  rpn_rank<<<dim3((NB*SORTN)/256), 256, 0, stream>>>(sidx, rankbg);

  rpn_labels2<<<dim3((NB*NA + 255)/256), 256, 0, stream>>>(labels, rankfg, rankbg, accI, accI + 2);
  rpn_targets<<<dim3((NB*NA + 255)/256), 256, 0, stream>>>(anch, gt_boxes, amax, tgt);
  rpn_loss_cls<<<dim3((NB*NA + 255)/256), 256, 0, stream>>>(clsS, labels, accF);
  rpn_loss_box<<<dim3((NB*NA + 255)/256), 256, 0, stream>>>(bbox, tgt, labels, accI + 2, accF);
  rpn_finalize<<<1, 1, 0, stream>>>(accF, out);
}
